// GPTModel_68788196212995
// MI455X (gfx1250) — matrix operand order repaired and checked by reading, not compiled, not run on hardware
//
#include <hip/hip_runtime.h>
#include <math.h>

// GPT forward on MI455X (gfx1250). Traffic analysis: fp32 weights ~650MB,
// K/V re-reads (shared 4-ways now) ~270MB/layer but L2-resident (qkv16=24MB
// << 192MB L2). All GEMM/attention math via v_wmma_f32_16x16x32_f16; LDS tile
// staging via Tensor Data Mover (tensor_load_to_lds) with double buffering so
// DMA overlaps WMMA; softmax reductions via v_permlane16 (VALU, not LDS).

#ifndef __has_builtin
#define __has_builtin(x) 0
#endif
#if __has_builtin(__builtin_amdgcn_tensor_load_to_lds) && \
    __has_builtin(__builtin_amdgcn_s_wait_tensorcnt)
#define HAVE_TDM 1
#else
#define HAVE_TDM 0
#endif

#define TT 2048
#define CC 1024
#define HH 16
#define DD 64
#define LL 8
#define VV 32000
#define BB 2
#define BT (BB * TT)
#define QW 4  // waves per attention workgroup (64 query rows share K/V tiles)

typedef __attribute__((ext_vector_type(16))) _Float16 v16h;
typedef __attribute__((ext_vector_type(8)))  float    v8f;

union Frag16 {
  v16h v;
  uint4 u[2];
  _Float16 h[16];
};

__device__ __forceinline__ v8f wmma_f16(v16h a, v16h b, v8f c) {
  return __builtin_amdgcn_wmma_f32_16x16x32_f16(false, a, false, b, (short)0, c,
                                                false, false);
}

// ---------------- Tensor Data Mover: 2-D f16 tile global -> LDS ------------
#if HAVE_TDM
typedef __attribute__((ext_vector_type(4))) unsigned int u32x4;
typedef __attribute__((ext_vector_type(8))) int i32x8;
typedef __attribute__((ext_vector_type(4))) int i32x4;

// D# per CDNA5 ISA 8.3/8.4: group0 = {count=1, lds_addr, global_addr, type=2},
// group1 = {data_size=2B, tensor_dim0/1 = tile dims (no OOB), tile_dim0/1,
// tensor_dim0_stride = row stride}. 2-D tensor: groups 2/3 zero.
__device__ __forceinline__ void tdm_load_2d_f16(const void* lds_dst,
                                                const void* global_src,
                                                unsigned tile_w, unsigned tile_h,
                                                unsigned row_stride_elems) {
  unsigned long long ga = (unsigned long long)global_src;
  unsigned lds = (unsigned)(unsigned long long)lds_dst;  // low 32 = LDS offset
  u32x4 g0 = {1u, lds, (unsigned)ga,
              (unsigned)((ga >> 32) & 0x01FFFFFFull) | 0x80000000u};
  i32x8 g1 = {(int)(1u << 16),        // data_size = 2 bytes
              (int)(tile_w << 16),    // tensor_dim0[15:0]
              (int)(tile_h << 16),    // tensor_dim1[15:0]
              (int)(tile_w << 16),    // tile_dim0
              (int)tile_h,            // tile_dim1 (tile_dim2 = 0)
              (int)row_stride_elems,  // tensor_dim0_stride[31:0]
              0, 0};
  i32x4 z4 = {0, 0, 0, 0};
#if __clang_major__ >= 23
  i32x8 z8 = {0, 0, 0, 0, 0, 0, 0, 0};
  __builtin_amdgcn_tensor_load_to_lds(g0, g1, z4, z4, z8, 0);
#else
  __builtin_amdgcn_tensor_load_to_lds(g0, g1, z4, z4, 0);
#endif
}
#define TDM_WAIT0() __builtin_amdgcn_s_wait_tensorcnt(0)
#else
#define TDM_WAIT0() ((void)0)
#endif

// ---------------- 16-lane-row XOR butterfly via v_permlane16 ---------------
template <int M, unsigned S0, unsigned S1>
__device__ __forceinline__ float xor16(float v) {
#if __has_builtin(__builtin_amdgcn_permlane16)
  unsigned s = __float_as_uint(v);
  unsigned r = __builtin_amdgcn_permlane16(s, s, S0, S1, false, false);
  return __uint_as_float(r);
#else
  return __shfl_xor(v, M, 32);
#endif
}
__device__ __forceinline__ float rmax16(float v) {
  v = fmaxf(v, xor16<1, 0x67452301u, 0xEFCDAB89u>(v));
  v = fmaxf(v, xor16<2, 0x54761032u, 0xDCFE98BAu>(v));
  v = fmaxf(v, xor16<4, 0x32107654u, 0xBA98FEDCu>(v));
  v = fmaxf(v, xor16<8, 0xFEDCBA98u, 0x76543210u>(v));
  return v;
}
__device__ __forceinline__ float rsum16(float v) {
  v += xor16<1, 0x67452301u, 0xEFCDAB89u>(v);
  v += xor16<2, 0x54761032u, 0xDCFE98BAu>(v);
  v += xor16<4, 0x32107654u, 0xBA98FEDCu>(v);
  v += xor16<8, 0xFEDCBA98u, 0x76543210u>(v);
  return v;
}

// -------------------------------------------------------------------------
// Embedding + sinusoidal positional encoding -> fp32 residual stream
// -------------------------------------------------------------------------
__global__ __launch_bounds__(256)
void embed_kernel(const int* __restrict__ tokens,
                  const float* __restrict__ emb, float* __restrict__ x) {
  const int row = blockIdx.x;  // b*T + t
  const int t = row & (TT - 1);
  const int tok = tokens[row];
  const float* er = emb + (size_t)tok * CC;
  float* xr = x + (size_t)row * CC;
  const float nlog = -logf(10000.0f) / (float)CC;
  for (int c = threadIdx.x; c < CC; c += 256) {
    int i2 = c & ~1;
    float ang = (float)t * __expf(nlog * (float)i2);
    float pe = (c & 1) ? __cosf(ang) : __sinf(ang);
    xr[c] = er[c] + pe;
  }
}

// -------------------------------------------------------------------------
// LayerNorm over C=1024, f16 output (WMMA A operand)
// -------------------------------------------------------------------------
__global__ __launch_bounds__(256)
void ln_kernel(const float* __restrict__ x, const float* __restrict__ w,
               const float* __restrict__ b, _Float16* __restrict__ out) {
  const int row = blockIdx.x, tid = threadIdx.x;
  const float* xr = x + (size_t)row * CC;
  float v[4], s = 0.f, s2 = 0.f;
#pragma unroll
  for (int i = 0; i < 4; ++i) {
    v[i] = xr[tid + 256 * i];
    s += v[i]; s2 += v[i] * v[i];
  }
#pragma unroll
  for (int m = 1; m < 32; m <<= 1) {
    s += __shfl_xor(s, m, 32);
    s2 += __shfl_xor(s2, m, 32);
  }
  __shared__ float sh[8], sh2[8];
  if ((tid & 31) == 0) { sh[tid >> 5] = s; sh2[tid >> 5] = s2; }
  __syncthreads();
  s = 0.f; s2 = 0.f;
#pragma unroll
  for (int i = 0; i < 8; ++i) { s += sh[i]; s2 += sh2[i]; }
  const float mu = s * (1.f / CC);
  const float var = s2 * (1.f / CC) - mu * mu;
  const float inv = rsqrtf(var + 1e-5f);
#pragma unroll
  for (int i = 0; i < 4; ++i) {
    int c = tid + 256 * i;
    out[(size_t)row * CC + c] = (_Float16)((v[i] - mu) * inv * w[c] + b[c]);
  }
}

// -------------------------------------------------------------------------
// Tiled WMMA GEMM, double-buffered LDS, A-tile via TDM, W fp32->f16 in regs.
// 256 threads = 8 waves; workgroup tile 128x128; wave tile 32x64 (2x4 accs).
// -------------------------------------------------------------------------
template <bool GELU_, bool RESID, bool OUTF16>
__global__ __launch_bounds__(256)
void gemm_kernel(const _Float16* __restrict__ A, const float* __restrict__ W,
                 const float* __restrict__ bias, const float* __restrict__ res,
                 float* __restrict__ outF, _Float16* __restrict__ outH,
                 int M, int N, int K) {
  __shared__ __align__(16) _Float16 As[2][128 * 32];  // [m][k] row-major
  __shared__ __align__(16) _Float16 Bt[2][128 * 32];  // [n][k] n-major
  const int tid = threadIdx.x;
  const int lane = tid & 31, w = tid >> 5;
  const int hf = lane >> 4, n16 = lane & 15;
  const int wrow = w >> 1, wcol = w & 1;
  const int row0 = blockIdx.y * 128;
  const int col0 = blockIdx.x * 128;
  const int kk = tid >> 3, nb = (tid & 7) * 16;  // B-tile load coords

  v8f acc[2][4];
#pragma unroll
  for (int mt = 0; mt < 2; ++mt)
#pragma unroll
    for (int nt = 0; nt < 4; ++nt)
#pragma unroll
      for (int j = 0; j < 8; ++j) acc[mt][nt][j] = 0.f;

  float4 br[4];
  const int nk = K >> 5;

  // ---- stage helpers (inline) ----
#define STAGE_A(buf, k0)                                                      \
  do {                                                                        \
    HAVE_TDM_A(buf, k0)                                                       \
  } while (0)
#if HAVE_TDM
#define HAVE_TDM_A(buf, k0)                                                   \
  if (tid < 32)                                                               \
    tdm_load_2d_f16(As[buf], A + (size_t)row0 * K + (k0), 32u, 128u,          \
                    (unsigned)K);
#else
#define HAVE_TDM_A(buf, k0)                                                   \
  _Pragma("unroll") for (int i = 0; i < 2; ++i) {                             \
    int u = tid * 2 + i;                                                      \
    int r = u >> 2, c = (u & 3) * 8;                                          \
    *(uint4*)(As[buf] + r * 32 + c) =                                         \
        *(const uint4*)(A + (size_t)(row0 + r) * K + (k0) + c);               \
  }
#endif
#define LOAD_B(k0)                                                            \
  do {                                                                        \
    const float* wp = W + (size_t)((k0) + kk) * N + col0 + nb;                \
    _Pragma("unroll") for (int i = 0; i < 4; ++i) br[i] =                     \
        ((const float4*)wp)[i];                                               \
  } while (0)
#define STORE_B(buf)                                                          \
  do {                                                                        \
    _Pragma("unroll") for (int i = 0; i < 4; ++i) {                           \
      Bt[buf][(nb + i * 4 + 0) * 32 + kk] = (_Float16)br[i].x;                \
      Bt[buf][(nb + i * 4 + 1) * 32 + kk] = (_Float16)br[i].y;                \
      Bt[buf][(nb + i * 4 + 2) * 32 + kk] = (_Float16)br[i].z;                \
      Bt[buf][(nb + i * 4 + 3) * 32 + kk] = (_Float16)br[i].w;                \
    }                                                                         \
  } while (0)

  // ---- prologue: fill buffer 0 ----
  STAGE_A(0, 0);
  LOAD_B(0);
  STORE_B(0);
#if HAVE_TDM
  if (tid < 32) TDM_WAIT0();
#endif
  __syncthreads();

  for (int kt = 0; kt < nk; ++kt) {
    const int cur = kt & 1;
    const bool more = (kt + 1) < nk;
    if (more) {  // kick next tile: TDM for A, global loads for W
      STAGE_A(cur ^ 1, (kt + 1) * 32);
      LOAD_B((kt + 1) * 32);
    }
    // compute on current buffer
    Frag16 af[2], bf[4];
#pragma unroll
    for (int mt = 0; mt < 2; ++mt) {  // A frag: K = {8h..}+{16+8h..}
      int m = wrow * 32 + mt * 16 + n16;
      af[mt].u[0] = *(const uint4*)(As[cur] + m * 32 + 8 * hf);
      af[mt].u[1] = *(const uint4*)(As[cur] + m * 32 + 16 + 8 * hf);
    }
#pragma unroll
    for (int nt = 0; nt < 4; ++nt) {  // B frag: K = {8h..}+{16+8h..}
      int n = wcol * 64 + nt * 16 + n16;
      bf[nt].u[0] = *(const uint4*)(Bt[cur] + n * 32 + 8 * hf);
      bf[nt].u[1] = *(const uint4*)(Bt[cur] + n * 32 + 16 + 8 * hf);
    }
#pragma unroll
    for (int mt = 0; mt < 2; ++mt)
#pragma unroll
      for (int nt = 0; nt < 4; ++nt)
        acc[mt][nt] = wmma_f16(af[mt].v, bf[nt].v, acc[mt][nt]);
    if (more) {
      STORE_B(cur ^ 1);
#if HAVE_TDM
      if (tid < 32) TDM_WAIT0();
#endif
    }
    __syncthreads();
  }
#undef STAGE_A
#undef HAVE_TDM_A
#undef LOAD_B
#undef STORE_B

  // epilogue: C/D layout row = j + 8*half, col = lane&15
#pragma unroll
  for (int mt = 0; mt < 2; ++mt) {
#pragma unroll
    for (int nt = 0; nt < 4; ++nt) {
      int ncol = col0 + wcol * 64 + nt * 16 + n16;
      float bv = bias ? bias[ncol] : 0.f;
#pragma unroll
      for (int j = 0; j < 8; ++j) {
        int r = row0 + wrow * 32 + mt * 16 + 8 * hf + j;
        float v = acc[mt][nt][j] + bv;
        if (GELU_) v = 0.5f * v * (1.f + erff(v * 0.70710678118f));
        size_t idx = (size_t)r * N + ncol;
        if (RESID)       outF[idx] = res[idx] + v;
        else if (OUTF16) outH[idx] = (_Float16)v;
        else             outF[idx] = v;
      }
    }
  }
}

// -------------------------------------------------------------------------
// Flash attention: 4 waves per WG share double-buffered K/V tiles (TDM).
// Wave w owns query rows [q0w + 16w, +16). qkv16 = [B*T, 3C] f16.
// -------------------------------------------------------------------------
__global__ __launch_bounds__(32 * QW)
void attn_kernel(const _Float16* __restrict__ qkv, _Float16* __restrict__ o) {
  const int tid = threadIdx.x;
  const int lane = tid & 31, w = tid >> 5;
  const int hf = lane >> 4, n16 = lane & 15;
  const int qb = blockIdx.x & (TT / 64 - 1);
  const int h = (blockIdx.x >> 5) & (HH - 1);
  const int b = blockIdx.x >> 9;
  const int q0w = qb * 64;
  const int q0 = q0w + w * 16;

  __shared__ __align__(16) _Float16 Ks[2][32 * DD];
  __shared__ __align__(16) _Float16 Vs[2][32 * DD];
  __shared__ __align__(16) _Float16 Ps[QW][16 * 32];

  // Q A-fragments (two 32-wide D chunks), held in registers
  Frag16 aq[2];
  {
    const _Float16* qr = qkv + (size_t)(b * TT + q0 + n16) * (3 * CC) + h * DD;
#pragma unroll
    for (int c0 = 0; c0 < 2; ++c0) {
      aq[c0].u[0] = *(const uint4*)(qr + c0 * 32 + 8 * hf);
      aq[c0].u[1] = *(const uint4*)(qr + c0 * 32 + 16 + 8 * hf);
    }
  }

  v8f accO[4];
  float mrun[8], lrun[8];
#pragma unroll
  for (int j = 0; j < 8; ++j) { mrun[j] = -1e30f; lrun[j] = 0.f; }
#pragma unroll
  for (int dt = 0; dt < 4; ++dt)
#pragma unroll
    for (int j = 0; j < 8; ++j) accO[dt][j] = 0.f;

#if HAVE_TDM
#define STAGE_KV(buf, k0)                                                     \
  if (tid < 32) {                                                             \
    const _Float16* kg =                                                      \
        qkv + (size_t)(b * TT + (k0)) * (3 * CC) + CC + h * DD;               \
    const _Float16* vg =                                                      \
        qkv + (size_t)(b * TT + (k0)) * (3 * CC) + 2 * CC + h * DD;           \
    tdm_load_2d_f16(Ks[buf], kg, (unsigned)DD, 32u, 3u * CC);                 \
    tdm_load_2d_f16(Vs[buf], vg, (unsigned)DD, 32u, 3u * CC);                 \
  }
#else
#define STAGE_KV(buf, k0)                                                     \
  do {                                                                        \
    int r = tid >> 2, sg = (tid & 3) * 16;                                    \
    const _Float16* kg =                                                      \
        qkv + (size_t)(b * TT + (k0) + r) * (3 * CC) + CC + h * DD + sg;      \
    const _Float16* vg =                                                      \
        qkv + (size_t)(b * TT + (k0) + r) * (3 * CC) + 2 * CC + h * DD + sg;  \
    ((uint4*)(Ks[buf] + r * DD + sg))[0] = ((const uint4*)kg)[0];             \
    ((uint4*)(Ks[buf] + r * DD + sg))[1] = ((const uint4*)kg)[1];             \
    ((uint4*)(Vs[buf] + r * DD + sg))[0] = ((const uint4*)vg)[0];             \
    ((uint4*)(Vs[buf] + r * DD + sg))[1] = ((const uint4*)vg)[1];             \
  } while (0)
#endif

  const int nkb = (q0w + 64 + 31) >> 5;  // key blocks for whole workgroup
  STAGE_KV(0, 0);
#if HAVE_TDM
  if (tid < 32) TDM_WAIT0();
#endif
  __syncthreads();

  for (int kb = 0; kb < nkb; ++kb) {
    const int cur = kb & 1;
    const int k0 = kb * 32;
    const bool more = (kb + 1) < nkb;
    if (more) STAGE_KV(cur ^ 1, k0 + 32);  // DMA next block under compute

    if (k0 <= q0 + 15) {  // wave-uniform causal guard (EXEC stays all-1s)
      // scores: two 16x16 tiles, 2 WMMAs each over D=64
      v8f sc[2];
#pragma unroll
      for (int st = 0; st < 2; ++st) {
#pragma unroll
        for (int j = 0; j < 8; ++j) sc[st][j] = 0.f;
#pragma unroll
        for (int c0 = 0; c0 < 2; ++c0) {
          Frag16 bk;
          const _Float16* kp =
              Ks[cur] + (st * 16 + n16) * DD + c0 * 32 + 8 * hf;
          bk.u[0] = *(const uint4*)kp;
          bk.u[1] = *(const uint4*)(kp + 16);
          sc[st] = wmma_f16(aq[c0].v, bk.v, sc[st]);
        }
      }
      // scale + causal mask + online softmax (row = j + 8*hf)
      _Float16* Pw = Ps[w];
#pragma unroll
      for (int j = 0; j < 8; ++j) {
        const int qg = q0 + j + 8 * hf;
        float v0 = sc[0][j] * 0.125f;
        float v1 = sc[1][j] * 0.125f;
        if (k0 + n16 > qg) v0 = -1e30f;
        if (k0 + 16 + n16 > qg) v1 = -1e30f;
        float nm = fmaxf(mrun[j], rmax16(fmaxf(v0, v1)));
        float p0 = __expf(v0 - nm), p1 = __expf(v1 - nm);
        float rs = rsum16(p0 + p1);
        float corr = __expf(mrun[j] - nm);
        lrun[j] = lrun[j] * corr + rs;
        mrun[j] = nm;
#pragma unroll
        for (int dt = 0; dt < 4; ++dt) accO[dt][j] *= corr;
        Pw[(j + 8 * hf) * 32 + n16] = (_Float16)p0;
        Pw[(j + 8 * hf) * 32 + 16 + n16] = (_Float16)p1;
      }
      // (compiler inserts s_wait_dscnt between aliasing Ps stores/loads)
      // P(16x32) @ V(32x64)
      Frag16 ap;
      ap.u[0] = *(const uint4*)(Pw + n16 * 32 + 16 * hf);
      ap.u[1] = *(const uint4*)(Pw + n16 * 32 + 16 * hf + 8);
#pragma unroll
      for (int dt = 0; dt < 4; ++dt) {
        Frag16 bv;
#pragma unroll
        for (int i = 0; i < 16; ++i)
          bv.h[i] = Vs[cur][(16 * hf + i) * DD + dt * 16 + n16];
        accO[dt] = wmma_f16(ap.v, bv.v, accO[dt]);
      }
    }
    if (more) {
#if HAVE_TDM
      if (tid < 32) TDM_WAIT0();
#endif
    }
    __syncthreads();
  }
#undef STAGE_KV

  // normalize + store O as f16 [B,T,C]
#pragma unroll
  for (int dt = 0; dt < 4; ++dt)
#pragma unroll
    for (int j = 0; j < 8; ++j) {
      int row = q0 + j + 8 * hf;
      int d = dt * 16 + n16;
      o[(size_t)(b * TT + row) * CC + h * DD + d] =
          (_Float16)(accO[dt][j] / lrun[j]);
    }
}

// -------------------------------------------------------------------------
extern "C" void kernel_launch(void* const* d_in, const int* in_sizes, int n_in,
                              void* d_out, int out_size, void* d_ws,
                              size_t ws_size, hipStream_t stream) {
  (void)in_sizes; (void)n_in; (void)out_size; (void)ws_size;
  const int*   tokens  = (const int*)  d_in[0];
  const float* tok_emb = (const float*)d_in[1];
  const float* qkv_w   = (const float*)d_in[2];
  const float* qkv_b   = (const float*)d_in[3];
  const float* proj_w  = (const float*)d_in[4];
  const float* proj_b  = (const float*)d_in[5];
  const float* fc1_w   = (const float*)d_in[6];
  const float* fc1_b   = (const float*)d_in[7];
  const float* fc2_w   = (const float*)d_in[8];
  const float* fc2_b   = (const float*)d_in[9];
  const float* ln1_w   = (const float*)d_in[10];
  const float* ln1_b   = (const float*)d_in[11];
  const float* ln2_w   = (const float*)d_in[12];
  const float* ln2_b   = (const float*)d_in[13];
  const float* lnf_w   = (const float*)d_in[14];
  const float* lnf_b   = (const float*)d_in[15];
  const float* head_w  = (const float*)d_in[16];

  // workspace layout (64 MB): qkv16 and FC1 activations share a region
  char* ws = (char*)d_ws;
  float*    x   = (float*)ws;                            // 16 MB residual
  _Float16* h16 = (_Float16*)(ws + (size_t)(16u << 20)); //  8 MB LN output
  _Float16* o16 = (_Float16*)(ws + (size_t)(24u << 20)); //  8 MB attn output
  _Float16* big = (_Float16*)(ws + (size_t)(32u << 20)); // 32 MB qkv16/act16
  _Float16* qkv16 = big;
  _Float16* a16   = big;

  embed_kernel<<<BT, 256, 0, stream>>>(tokens, tok_emb, x);

  for (int l = 0; l < LL; ++l) {
    ln_kernel<<<BT, 256, 0, stream>>>(x, ln1_w + l * CC, ln1_b + l * CC, h16);
    gemm_kernel<false, false, true>
        <<<dim3(3 * CC / 128, BT / 128), 256, 0, stream>>>(
            h16, qkv_w + (size_t)l * CC * 3 * CC, qkv_b + l * 3 * CC, nullptr,
            nullptr, qkv16, BT, 3 * CC, CC);
    attn_kernel<<<BB * HH * (TT / 64), 32 * QW, 0, stream>>>(qkv16, o16);
    gemm_kernel<false, true, false>
        <<<dim3(CC / 128, BT / 128), 256, 0, stream>>>(
            o16, proj_w + (size_t)l * CC * CC, proj_b + l * CC, x, x, nullptr,
            BT, CC, CC);
    ln_kernel<<<BT, 256, 0, stream>>>(x, ln2_w + l * CC, ln2_b + l * CC, h16);
    gemm_kernel<true, false, true>
        <<<dim3(4 * CC / 128, BT / 128), 256, 0, stream>>>(
            h16, fc1_w + (size_t)l * CC * 4 * CC, fc1_b + l * 4 * CC, nullptr,
            nullptr, a16, BT, 4 * CC, CC);
    gemm_kernel<false, true, false>
        <<<dim3(CC / 128, BT / 128), 256, 0, stream>>>(
            a16, fc2_w + (size_t)l * 4 * CC * CC, fc2_b + l * CC, x, x, nullptr,
            BT, CC, 4 * CC);
  }

  ln_kernel<<<BT, 256, 0, stream>>>(x, lnf_w, lnf_b, h16);
  gemm_kernel<false, false, false>
      <<<dim3(VV / 128, BT / 128), 256, 0, stream>>>(
          h16, head_w, nullptr, nullptr, (float*)d_out, nullptr, BT, VV, CC);
}
